// SelfAttention_43877385896485
// MI455X (gfx1250) — matrix operand order repaired and checked by reading, not compiled, not run on hardware
//
#include <hip/hip_runtime.h>
#include <hip/hip_bf16.h>
#include <cstdint>

// Problem sizes (fixed by the reference)
#define S_LEN 4096
#define D_DIM 2048

typedef __attribute__((ext_vector_type(16))) __bf16         v16bf;
typedef __attribute__((ext_vector_type(8)))  float          v8f;
typedef __attribute__((ext_vector_type(8)))  unsigned int   v8u;
typedef __attribute__((ext_vector_type(4)))  float          v4f;
typedef __attribute__((ext_vector_type(4)))  unsigned short v4us;

__device__ __forceinline__ unsigned short f32_bf16(float f) {
  unsigned int u = __builtin_bit_cast(unsigned int, f);
  u += 0x7FFFu + ((u >> 16) & 1u);            // round-to-nearest-even
  return (unsigned short)(u >> 16);
}

// ---- WMMA fragment gathers from LDS (stride in ushorts) --------------------
// A fragment: 16x32 bf16 (ISA 7.12.2): lane L row M=L%16; VGPR r holds
// K = (r&3)*2 + (r>>2)*16 + (L>=16 ? 8 : 0), two consecutive K per VGPR.
__device__ __forceinline__ v16bf load_frag_a(const unsigned short* p, int lane, int stride) {
  v8u au;
  const int m  = lane & 15;
  const int kb = (lane >> 4) * 8;
#pragma unroll
  for (int r = 0; r < 8; ++r) {
    const int kk = (r & 3) * 2 + (r >> 2) * 16 + kb;
    au[r] = *(const unsigned int*)(p + m * stride + kk);
  }
  return __builtin_bit_cast(v16bf, au);
}
// B fragment: 32x16 bf16 (KxN), LDS stored [n][k]: lane L col N=L%16; VGPR r
// holds K = (r&3)*2 + (r>>2)*16 + (L>=16 ? 8 : 0), two consecutive K per VGPR.
__device__ __forceinline__ v16bf load_frag_b(const unsigned short* p, int lane, int stride) {
  v8u bu;
  const int n  = lane & 15;
  const int kb = (lane >> 4) * 8;
#pragma unroll
  for (int r = 0; r < 8; ++r) {
    bu[r] = *(const unsigned int*)(p + n * stride + kb + (r & 3) * 2 + (r >> 2) * 16);
  }
  return __builtin_bit_cast(v16bf, bu);
}

// ---- CDNA5 async global->LDS copy (ASYNCcnt path) --------------------------
__device__ __forceinline__ unsigned lds_off(const void* p) {
  // flat LDS addresses carry the WG-relative offset in the low 32 bits
  return (unsigned)(unsigned long long)p;
}
__device__ __forceinline__ void async_b128(unsigned ldsaddr, unsigned goff, const void* base) {
  asm volatile("global_load_async_to_lds_b128 %0, %1, %2"
               :: "v"(ldsaddr), "v"(goff), "s"(base) : "memory");
}
__device__ __forceinline__ void async_wait0() {
  asm volatile("s_wait_asynccnt 0x0" ::: "memory");
}

// ===========================================================================
// Shared 128x128 bf16 NT-GEMM pipeline, BK=64, async double-buffered LDS.
// A[m][k] row-major, B stored [n][k]; 256 threads; wave w -> 32x64 strip
// (wr = w>>1 row strip, wc = w&1 col strip); 16 WMMA per wave per chunk.
// ===========================================================================
#define ROWU  136                 // LDS row stride in ushorts (272B, bank-safe)
#define TILEU (128 * ROWU)        // ushorts per tile buffer

__device__ __forceinline__ v8f wmma_bf16(v16bf a, v16bf b, v8f c) {
  return __builtin_amdgcn_wmma_f32_16x16x32_bf16(false, a, false, b, (short)0, c, false, false);
}

__device__ __forceinline__ void mm_tile(const unsigned short* __restrict__ Ag, unsigned aRow0,
                                        const unsigned short* __restrict__ Bg, unsigned bRow0,
                                        unsigned kElems, int NK, int tid,
                                        unsigned short* ldsA, unsigned short* ldsB,
                                        v8f (&acc)[8]) {
  const int lane = tid & 31;
  const int wave = tid >> 5;
  const int wr = wave >> 1;
  const int wc = wave & 1;
  const unsigned ldsA0 = lds_off(ldsA);
  const unsigned ldsB0 = lds_off(ldsB);

  auto issue = [&](int kc, int buf) {
    // tile = 128 rows x 256B payload = 2048 x 16B chunks; 8 per thread per tile
#pragma unroll
    for (int i = 0; i < 8; ++i) {
      const int ch  = i * 256 + tid;
      const int row = ch >> 4, c16 = ch & 15;
      const unsigned lrel = (unsigned)(buf * TILEU + row * ROWU) * 2u + c16 * 16u;
      const unsigned ga = ((aRow0 + row) * kElems + (unsigned)kc * 64u) * 2u + c16 * 16u;
      const unsigned gb = ((bRow0 + row) * kElems + (unsigned)kc * 64u) * 2u + c16 * 16u;
      async_b128(ldsA0 + lrel, ga, Ag);
      async_b128(ldsB0 + lrel, gb, Bg);
    }
  };

  issue(0, 0);
  async_wait0();
  __syncthreads();

  for (int k = 0; k < NK; ++k) {
    if (k + 1 < NK) issue(k + 1, (k + 1) & 1);
    const unsigned short* bA = ldsA + (k & 1) * TILEU + (wr * 32) * ROWU;
    const unsigned short* bB = ldsB + (k & 1) * TILEU + (wc * 64) * ROWU;
#pragma unroll
    for (int s = 0; s < 2; ++s) {                 // two 32-wide sub-chunks
      const v16bf a0 = load_frag_a(bA + s * 64, lane, ROWU);
      const v16bf a1 = load_frag_a(bA + 16 * ROWU + s * 64, lane, ROWU);
#pragma unroll
      for (int c = 0; c < 4; ++c) {
        const v16bf b = load_frag_b(bB + c * 16 * ROWU + s * 64, lane, ROWU);
        acc[c]     = wmma_bf16(a0, b, acc[c]);
        acc[4 + c] = wmma_bf16(a1, b, acc[4 + c]);
      }
    }
    async_wait0();
    __syncthreads();
  }
}

// ===========================================================================
// Stage 0a: elementwise fp32 -> bf16 (embeddings).  n must be /4.
// ===========================================================================
__global__ __launch_bounds__(256) void k_cvt(const float* __restrict__ in,
                                             unsigned short* __restrict__ out) {
  const size_t i = ((size_t)blockIdx.x * 256 + threadIdx.x) * 4;
  const v4f x = *(const v4f*)(in + i);
  v4us o;
#pragma unroll
  for (int j = 0; j < 4; ++j) o[j] = f32_bf16(x[j]);
  *(v4us*)(out + i) = o;
}

// ===========================================================================
// Stage 0b: transpose-convert W[k][n] fp32 -> Wt[n][k] bf16 (64x64 LDS tiles).
// ===========================================================================
__global__ __launch_bounds__(256) void k_cvt_t(const float* __restrict__ W,
                                               unsigned short* __restrict__ Wt) {
  __shared__ unsigned short t[64][65];
  const int n0 = blockIdx.x * 64, k0 = blockIdx.y * 64;
  const int tid = threadIdx.x;
#pragma unroll
  for (int i = 0; i < 16; ++i) {
    const int e = i * 256 + tid;
    const int k = e >> 6, n = e & 63;           // coalesced read over n
    t[n][k] = f32_bf16(W[(size_t)(k0 + k) * D_DIM + (n0 + n)]);
  }
  __syncthreads();
#pragma unroll
  for (int i = 0; i < 16; ++i) {
    const int e = i * 256 + tid;
    const int n = e >> 6, k = e & 63;           // coalesced write over k
    Wt[(size_t)(n0 + n) * D_DIM + (k0 + k)] = t[n][k];
  }
}

// ===========================================================================
// Stage 1: C = X[bf16, SxD] @ W^T-layout[bf16, DxD]; TR stores C^T (for V).
// ===========================================================================
template <bool TR>
__global__ __launch_bounds__(256) void k_qkv(const unsigned short* __restrict__ X,
                                             const unsigned short* __restrict__ Wt,
                                             unsigned short* __restrict__ out) {
  __shared__ unsigned short ldsA[2 * TILEU];
  __shared__ unsigned short ldsB[2 * TILEU];
  const int tid = threadIdx.x;
  const int bn0 = blockIdx.x * 128;
  const int bm0 = blockIdx.y * 128;
  v8f acc[8] = {};
  mm_tile(X, bm0, Wt, bn0, D_DIM, D_DIM / 64, tid, ldsA, ldsB, acc);

  const int lane = tid & 31, wave = tid >> 5;
  const int rbase = bm0 + (wave >> 1) * 32 + (lane >> 4) * 8;
  const int cbase = bn0 + (wave & 1) * 64 + (lane & 15);
#pragma unroll
  for (int a = 0; a < 2; ++a)
#pragma unroll
    for (int c = 0; c < 4; ++c)
#pragma unroll
      for (int r = 0; r < 8; ++r) {
        const int row = rbase + a * 16 + r;
        const int col = cbase + c * 16;
        if (TR) out[(size_t)col * S_LEN + row] = f32_bf16(acc[a * 4 + c][r]);
        else    out[(size_t)row * D_DIM + col] = f32_bf16(acc[a * 4 + c][r]);
      }
}

// ===========================================================================
// Stage 2: scores[f32, SxS] = (Q @ K^T)/10.  Causal masking is analytic in
// softmax; fully masked tiles exit before touching memory.
// ===========================================================================
__global__ __launch_bounds__(256) void k_scores(const unsigned short* __restrict__ Q,
                                                const unsigned short* __restrict__ K,
                                                float* __restrict__ scores,
                                                const int* __restrict__ inc_mask) {
  const int causal = (*inc_mask == 0);
  const int bj0 = blockIdx.x * 128;   // key block (columns)
  const int bi0 = blockIdx.y * 128;   // query block (rows)
  if (causal && bj0 > bi0 + 127) return;

  __shared__ unsigned short ldsA[2 * TILEU];
  __shared__ unsigned short ldsB[2 * TILEU];
  const int tid = threadIdx.x;
  v8f acc[8] = {};
  mm_tile(Q, bi0, K, bj0, D_DIM, D_DIM / 64, tid, ldsA, ldsB, acc);

  const int lane = tid & 31, wave = tid >> 5;
  const int rbase = bi0 + (wave >> 1) * 32 + (lane >> 4) * 8;
  const int cbase = bj0 + (wave & 1) * 64 + (lane & 15);
#pragma unroll
  for (int a = 0; a < 2; ++a)
#pragma unroll
    for (int c = 0; c < 4; ++c)
#pragma unroll
      for (int r = 0; r < 8; ++r)
        scores[(size_t)(rbase + a * 16 + r) * S_LEN + (cbase + c * 16)] =
            acc[a * 4 + c][r] * 0.1f;           // reference hardcodes /10
}

// ===========================================================================
// Stage 3: row softmax, fp32 in -> bf16 probs out.  Mask via loop bound
// (j <= row); zero-fill up to the row's 128-block end for the PV diagonal.
// ===========================================================================
__global__ __launch_bounds__(256) void k_softmax(const float* __restrict__ Sc,
                                                 unsigned short* __restrict__ Pb,
                                                 const int* __restrict__ inc_mask) {
  __shared__ float red[256];
  const int causal = (*inc_mask == 0);
  const int row = blockIdx.x;
  const int tid = threadIdx.x;
  const float* s = Sc + (size_t)row * S_LEN;
  unsigned short* p = Pb + (size_t)row * S_LEN;
  const int n    = causal ? (row + 1) : S_LEN;               // valid prefix
  const int wlim = causal ? ((row / 128) + 1) * 128 : S_LEN; // written prefix

  float m = -__builtin_inff();
  for (int i = tid; i < n; i += 256) m = fmaxf(m, s[i]);
  red[tid] = m;
  __syncthreads();
  for (int st = 128; st > 0; st >>= 1) {
    if (tid < st) red[tid] = fmaxf(red[tid], red[tid + st]);
    __syncthreads();
  }
  m = red[0];
  __syncthreads();

  float sum = 0.0f;
  for (int i = tid; i < n; i += 256) sum += __expf(s[i] - m);
  red[tid] = sum;
  __syncthreads();
  for (int st = 128; st > 0; st >>= 1) {
    if (tid < st) red[tid] += red[tid + st];
    __syncthreads();
  }
  const float inv = 1.0f / red[0];

  for (int i = tid; i < wlim; i += 256) {
    const float e = (i < n) ? __expf(s[i] - m) * inv : 0.0f;
    p[i] = f32_bf16(e);
  }
}

// ===========================================================================
// Stage 4: out[f32, SxD] = P[bf16, SxS] @ V (via V^T[bf16, DxS]); causal
// stops contraction at the row-block diagonal.
// ===========================================================================
__global__ __launch_bounds__(256) void k_pv(const unsigned short* __restrict__ Pb,
                                            const unsigned short* __restrict__ Vt,
                                            float* __restrict__ out,
                                            const int* __restrict__ inc_mask) {
  __shared__ unsigned short ldsA[2 * TILEU];
  __shared__ unsigned short ldsB[2 * TILEU];
  const int causal = (*inc_mask == 0);
  const int tid = threadIdx.x;
  const int bn0 = blockIdx.x * 128;   // output columns (d)
  const int bm0 = blockIdx.y * 128;   // output rows
  const int NK  = (causal ? (bm0 + 128) : S_LEN) / 64;

  v8f acc[8] = {};
  mm_tile(Pb, bm0, Vt, bn0, S_LEN, NK, tid, ldsA, ldsB, acc);

  const int lane = tid & 31, wave = tid >> 5;
  const int rbase = bm0 + (wave >> 1) * 32 + (lane >> 4) * 8;
  const int cbase = bn0 + (wave & 1) * 64 + (lane & 15);
#pragma unroll
  for (int a = 0; a < 2; ++a)
#pragma unroll
    for (int c = 0; c < 4; ++c)
#pragma unroll
      for (int r = 0; r < 8; ++r)
        out[(size_t)(rbase + a * 16 + r) * D_DIM + (cbase + c * 16)] = acc[a * 4 + c][r];
}

// ---------------------------------------------------------------------------
extern "C" void kernel_launch(void* const* d_in, const int* in_sizes, int n_in,
                              void* d_out, int out_size, void* d_ws, size_t ws_size,
                              hipStream_t stream) {
  const float* inp = (const float*)d_in[0];   // input_embedding  [S,D]
  const float* qry = (const float*)d_in[1];   // query_embedding  [S,D]
  const float* Wq  = (const float*)d_in[2];
  const float* Wk  = (const float*)d_in[3];
  const float* Wv  = (const float*)d_in[4];
  const int* inc_mask = (const int*)d_in[5];
  float* out = (float*)d_out;

  // Workspace (144 MB), with aliasing:
  //   [0..56MB)   : Xq, Xi bf16 [S,D] (16 MB ea) + Wqt, Wkt, Wvt bf16 [D,D] (8 MB ea)
  //   [0..64MB)   : Sc f32 [S,S]  (reuses the region above once stage 1 is done)
  //   [64..112MB) : Qb, Kb [S,D] bf16, Vt [D,S] bf16
  //   [112..144MB): Pb [S,S] bf16
  char* ws = (char*)d_ws;
  const size_t MB = 1024 * 1024;
  unsigned short* Xq  = (unsigned short*)(ws);
  unsigned short* Xi  = (unsigned short*)(ws + 16 * MB);
  unsigned short* Wqt = (unsigned short*)(ws + 32 * MB);
  unsigned short* Wkt = (unsigned short*)(ws + 40 * MB);
  unsigned short* Wvt = (unsigned short*)(ws + 48 * MB);
  float*          Sc  = (float*)(ws);                    // aliases the above
  unsigned short* Qb  = (unsigned short*)(ws + 64 * MB);
  unsigned short* Kb  = (unsigned short*)(ws + 80 * MB);
  unsigned short* Vt  = (unsigned short*)(ws + 96 * MB);
  unsigned short* Pb  = (unsigned short*)(ws + 112 * MB);

  const int nEmb = S_LEN * D_DIM;               // 8M elements
  const dim3 gCVT(nEmb / (256 * 4));
  const dim3 gCVTT(D_DIM / 64, D_DIM / 64);
  const dim3 gQKV(D_DIM / 128, S_LEN / 128);    // 16 x 32
  const dim3 gSC(S_LEN / 128, S_LEN / 128);     // 32 x 32
  const dim3 gPV(D_DIM / 128, S_LEN / 128);     // 16 x 32

  hipLaunchKernelGGL(k_cvt,   gCVT,  dim3(256), 0, stream, qry, Xq);
  hipLaunchKernelGGL(k_cvt,   gCVT,  dim3(256), 0, stream, inp, Xi);
  hipLaunchKernelGGL(k_cvt_t, gCVTT, dim3(256), 0, stream, Wq, Wqt);
  hipLaunchKernelGGL(k_cvt_t, gCVTT, dim3(256), 0, stream, Wk, Wkt);
  hipLaunchKernelGGL(k_cvt_t, gCVTT, dim3(256), 0, stream, Wv, Wvt);
  hipLaunchKernelGGL((k_qkv<false>), gQKV, dim3(256), 0, stream, Xq, Wqt, Qb);
  hipLaunchKernelGGL((k_qkv<false>), gQKV, dim3(256), 0, stream, Xi, Wkt, Kb);
  hipLaunchKernelGGL((k_qkv<true>),  gQKV, dim3(256), 0, stream, Xi, Wvt, Vt);
  hipLaunchKernelGGL(k_scores,  gSC, dim3(256), 0, stream, Qb, Kb, Sc, inc_mask);
  hipLaunchKernelGGL(k_softmax, dim3(S_LEN), dim3(256), 0, stream, Sc, Pb, inc_mask);
  hipLaunchKernelGGL(k_pv,      gPV, dim3(256), 0, stream, Pb, Vt, out, inc_mask);
}
